// GAT_65910568124592
// MI455X (gfx1250) — matrix operand order repaired and checked by reading, compiled, not run on hardware
//
#include <hip/hip_runtime.h>
#include <hip/hip_bf16.h>

// ---------------- problem constants (match reference) ----------------
#define N_NODES   50000
#define NPAD      50016          // N rounded up to multiple of 32
#define D         128            // d_model == hidden
#define E0        800000         // raw edges
#define E_TOT     850000         // + self loops
#define NGRAPH    512
#define NEG_SLOPE 0.2f

typedef __attribute__((ext_vector_type(16))) _Float16 v16h;
typedef __attribute__((ext_vector_type(8)))  _Float16 v8h;
typedef __attribute__((ext_vector_type(8)))  float    v8f;
typedef __attribute__((ext_vector_type(4)))  unsigned v4u;
typedef __attribute__((ext_vector_type(8)))  unsigned v8u;

union V16 { v16h v; v8h h[2]; };

// ---------------- helpers ----------------
__device__ __forceinline__ unsigned ford(float f) {
    unsigned u = __float_as_uint(f);
    return (u & 0x80000000u) ? ~u : (u | 0x80000000u);   // monotonic encoding
}
__device__ __forceinline__ float finv(unsigned u) {
    return (u & 0x80000000u) ? __uint_as_float(u & 0x7FFFFFFFu)
                             : __uint_as_float(~u);
}
#define MU_NEG_INF 0x007FFFFFu   // ford(-inf)

__device__ __forceinline__ float lrelu(float x) {
    return x > 0.f ? x : NEG_SLOPE * x;
}

// ---------------- kernels ----------------

// x_f16[n,f] = emb[node_ids[n], f]   (padded rows -> 0)
__global__ __launch_bounds__(256) void k_embed(const int* __restrict__ ids,
                                               const float* __restrict__ emb,
                                               _Float16* __restrict__ xh) {
    int t = blockIdx.x * 256 + threadIdx.x;
    if (t >= NPAD * D) return;
    int node = t >> 7, f = t & (D - 1);
    float v = (node < N_NODES) ? emb[(size_t)ids[node] * D + f] : 0.f;
    xh[t] = (_Float16)v;
}

// pack W (f32 row-major KxN, 128x128) into WMMA-B fragment order, f16.
// Wp[((n0*4+k0)*32 + lane)*16 + j] = W[k0*32 + (lane>>4)*16 + j][n0*16 + (lane&15)]
__global__ __launch_bounds__(256) void k_packW(const float* __restrict__ W,
                                               _Float16* __restrict__ Wp) {
    int t = blockIdx.x * 256 + threadIdx.x;
    if (t >= D * D) return;
    int j    = t & 15;
    int lane = (t >> 4) & 31;
    int k0   = (t >> 9) & 3;
    int n0   = (t >> 11) & 7;
    int k = k0 * 32 + (lane >> 4) * 16 + j;
    int n = n0 * 16 + (lane & 15);
    Wp[t] = (_Float16)W[(size_t)k * D + n];
}

// per-layer state init: acc = 0, m = -inf(encoded), s = 0
__global__ __launch_bounds__(256) void k_init(float* __restrict__ acc,
                                              unsigned* __restrict__ mu,
                                              float* __restrict__ s) {
    int t = blockIdx.x * 256 + threadIdx.x;
    if (t < NPAD * D) acc[t] = 0.f;
    if (t < NPAD) { mu[t] = MU_NEG_INF; s[t] = 0.f; }
}

// h = x @ W  via v_wmma_f32_16x16x32_f16.  One wave -> 16 rows x all 128 cols.
// W (packed, 32KB f16) is DMA'd into LDS by the Tensor Data Mover (one TDM
// descriptor issued by wave 0, completion via TENSORcnt), not bounced via VGPRs.
__global__ __launch_bounds__(256) void k_gemm(const _Float16* __restrict__ xh,
                                              const _Float16* __restrict__ Wp,
                                              float* __restrict__ h) {
    __shared__ _Float16 sW[D * D];                    // 32 KB packed B

    if (threadIdx.x == 0) {
        // ---- Tensor DMA Descriptor (ISA cdna5 ch.8) ----
        unsigned long long ga = (unsigned long long)Wp;          // global addr
        unsigned lds_off = (unsigned)(uintptr_t)(void*)sW;       // LDS byte offset
        // group0: count=1 | lds_addr | global_addr[56:0] | type=2
        v4u g0;
        g0.x = 1u;                                   // count=1, user mode
        g0.y = lds_off;                              // lds_addr
        g0.z = (unsigned)ga;                         // global_addr[31:0]
        g0.w = ((unsigned)(ga >> 32) & 0x01FFFFFFu)  // global_addr[56:32]
               | (2u << 30);                         // type=2 ("image")
        // group1: data_size=2B; 1 row of 16384 elements (32KB contiguous)
        v8u g1;
        g1.s0 = 0x00010000u;        // workgroup_mask=0, data_size=1 (2 bytes)
        g1.s1 = 16384u << 16;       // tensor_dim0[15:0]=16384 (bits 79:48)
        g1.s2 = 1u << 16;           // tensor_dim1=1 (bits 111:80)
        g1.s3 = 16384u << 16;       // tile_dim0=16384 (bits 127:112)
        g1.s4 = 1u;                 // tile_dim1=1
        g1.s5 = 16384u;             // tensor_dim0_stride low 32
        g1.s6 = 0u;
        g1.s7 = 0u;
        asm volatile("tensor_load_to_lds %0, %1"
                     :: "s"(g0), "s"(g1) : "memory");
        __builtin_amdgcn_s_wait_tensorcnt(0);
    }
    __syncthreads();

    const int wave = threadIdx.x >> 5, lane = threadIdx.x & 31;
    const int m0 = (blockIdx.x * 8 + wave) * 16;
    if (m0 >= NPAD) return;                           // wave-uniform: EXEC stays full
    const int hsel = lane >> 4, mr = lane & 15;
    const _Float16* arow = xh + (size_t)(m0 + mr) * D;

    v8f c[8] = {};
#pragma unroll
    for (int k0 = 0; k0 < 4; ++k0) {
        V16 a;
        a.h[0] = *(const v8h*)(arow + k0 * 32 + hsel * 16);       // K = 16*hsel..+7
        a.h[1] = *(const v8h*)(arow + k0 * 32 + hsel * 16 + 8);   // K = 16*hsel+8..+15
        V16 b[8];
#pragma unroll
        for (int n0 = 0; n0 < 8; ++n0) {              // burst all B fragments
            const _Float16* bp = sW + ((size_t)(n0 * 4 + k0) * 32 + lane) * 16;
            b[n0].h[0] = *(const v8h*)(bp);
            b[n0].h[1] = *(const v8h*)(bp + 8);
        }
#pragma unroll
        for (int n0 = 0; n0 < 8; ++n0)                // back-to-back WMMAs
            c[n0] = __builtin_amdgcn_wmma_f32_16x16x32_f16(
                false, a.v, false, b[n0].v, (short)0, c[n0], false, false);
    }
#pragma unroll
    for (int n0 = 0; n0 < 8; ++n0)
#pragma unroll
        for (int v = 0; v < 8; ++v) {
            int row = m0 + v + 8 * hsel;              // C/D layout (ISA 7.12.2)
            h[(size_t)row * D + n0 * 16 + mr] = c[n0][v];
        }
}

// per-node attention coefficients: as_[n] = h[n]·a_src, ad_[n] = h[n]·a_dst
__global__ __launch_bounds__(256) void k_att(const float* __restrict__ h,
                                             const float* __restrict__ a_s,
                                             const float* __restrict__ a_d,
                                             float* __restrict__ os,
                                             float* __restrict__ od) {
    int wave = threadIdx.x >> 5, lane = threadIdx.x & 31;
    int node = blockIdx.x * 8 + wave;
    if (node >= N_NODES) return;
    float4 hv = ((const float4*)(h + (size_t)node * D))[lane];
    float4 s4 = ((const float4*)a_s)[lane];
    float4 d4 = ((const float4*)a_d)[lane];
    float vs = hv.x * s4.x + hv.y * s4.y + hv.z * s4.z + hv.w * s4.w;
    float vd = hv.x * d4.x + hv.y * d4.y + hv.z * d4.z + hv.w * d4.w;
#pragma unroll
    for (int off = 16; off; off >>= 1) {
        vs += __shfl_xor(vs, off, 32);
        vd += __shfl_xor(vd, off, 32);
    }
    if (lane == 0) { os[node] = vs; od[node] = vd; }
}

// pass 1: e = leaky_relu(as[src]+ad[dst]); segment max into mu[dst]
__global__ __launch_bounds__(256) void k_emax(const int* __restrict__ src,
                                              const int* __restrict__ dst,
                                              const float* __restrict__ as_,
                                              const float* __restrict__ ad_,
                                              float* __restrict__ e,
                                              unsigned* __restrict__ mu) {
    long t = (long)blockIdx.x * 256 + threadIdx.x;
    if (t >= E_TOT) return;
    int s, d;
    if (t < E0) { s = src[t]; d = dst[t]; } else { s = d = (int)(t - E0); }
    float ev = lrelu(as_[s] + ad_[d]);
    e[t] = ev;
    atomicMax(&mu[d], ford(ev));
}

// pass 2: ee = exp(e - m[dst]); segment sum into s[dst]
__global__ __launch_bounds__(256) void k_esum(const int* __restrict__ src,
                                              const int* __restrict__ dst,
                                              const float* __restrict__ e,
                                              const unsigned* __restrict__ mu,
                                              float* __restrict__ ee,
                                              float* __restrict__ sums) {
    long t = (long)blockIdx.x * 256 + threadIdx.x;
    if (t >= E_TOT) return;
    int d = (t < E0) ? dst[t] : (int)(t - E0);
    float v = __expf(e[t] - finv(mu[d]));
    ee[t] = v;
    atomicAdd(&sums[d], v);
}

// pass 3: acc[dst] += (ee/(s+eps)) * h[src]   (wave per edge, float4 per lane)
__global__ __launch_bounds__(256) void k_aggr(const int* __restrict__ src,
                                              const int* __restrict__ dst,
                                              const float* __restrict__ ee,
                                              const float* __restrict__ sums,
                                              const float* __restrict__ h,
                                              float* __restrict__ acc) {
    int wave = threadIdx.x >> 5, lane = threadIdx.x & 31;
    long eid = (long)blockIdx.x * 8 + wave;
    if (eid >= E_TOT) return;
    int s, d;
    if (eid < E0) { s = src[eid]; d = dst[eid]; } else { s = d = (int)(eid - E0); }
    float alpha = ee[eid] / (sums[d] + 1e-16f);
    float4 hv = ((const float4*)(h + (size_t)s * D))[lane];
    float* ap = acc + (size_t)d * D + lane * 4;
    atomicAdd(ap + 0, alpha * hv.x);
    atomicAdd(ap + 1, alpha * hv.y);
    atomicAdd(ap + 2, alpha * hv.z);
    atomicAdd(ap + 3, alpha * hv.w);
}

// bias (+ optional relu); acc updated in place (fp32) + next-layer f16 input
__global__ __launch_bounds__(256) void k_finish(float* __restrict__ acc,
                                                const float* __restrict__ bias,
                                                _Float16* __restrict__ xh,
                                                int do_relu) {
    int t = blockIdx.x * 256 + threadIdx.x;
    if (t >= NPAD * D) return;
    int node = t >> 7, f = t & (D - 1);
    float v = (node < N_NODES) ? acc[t] + bias[f] : 0.f;
    if (do_relu) v = fmaxf(v, 0.f);
    acc[t] = v;
    xh[t] = (_Float16)v;
}

__global__ __launch_bounds__(256) void k_pool_zero(float* __restrict__ out,
                                                   float* __restrict__ cnt) {
    int t = blockIdx.x * 256 + threadIdx.x;
    if (t < NGRAPH * D) out[t] = 0.f;
    if (t < NGRAPH) cnt[t] = 0.f;
}

__global__ __launch_bounds__(256) void k_pool_sum(const float* __restrict__ x,
                                                  const int* __restrict__ batch,
                                                  float* __restrict__ out,
                                                  float* __restrict__ cnt) {
    int wave = threadIdx.x >> 5, lane = threadIdx.x & 31;
    int node = blockIdx.x * 8 + wave;
    if (node >= N_NODES) return;
    int g = batch[node];
    float4 v = ((const float4*)(x + (size_t)node * D))[lane];
    float* op = out + (size_t)g * D + lane * 4;
    atomicAdd(op + 0, v.x);
    atomicAdd(op + 1, v.y);
    atomicAdd(op + 2, v.z);
    atomicAdd(op + 3, v.w);
    if (lane == 0) atomicAdd(&cnt[g], 1.f);
}

__global__ __launch_bounds__(256) void k_pool_div(float* __restrict__ out,
                                                  const float* __restrict__ cnt) {
    int t = blockIdx.x * 256 + threadIdx.x;
    if (t >= NGRAPH * D) return;
    out[t] /= fmaxf(cnt[t >> 7], 1.f);
}

// ---------------- host side ----------------
extern "C" void kernel_launch(void* const* d_in, const int* in_sizes, int n_in,
                              void* d_out, int out_size, void* d_ws, size_t ws_size,
                              hipStream_t stream) {
    const int*   node_ids = (const int*)d_in[0];
    const int*   eidx     = (const int*)d_in[1];
    const int*   esrc     = eidx;
    const int*   edst     = eidx + E0;
    const int*   batch    = (const int*)d_in[2];
    const float* emb      = (const float*)d_in[3];
    const float* W[3]  = { (const float*)d_in[4],  (const float*)d_in[8],  (const float*)d_in[12] };
    const float* aS[3] = { (const float*)d_in[5],  (const float*)d_in[9],  (const float*)d_in[13] };
    const float* aD[3] = { (const float*)d_in[6],  (const float*)d_in[10], (const float*)d_in[14] };
    const float* B[3]  = { (const float*)d_in[7],  (const float*)d_in[11], (const float*)d_in[15] };
    float* out = (float*)d_out;

    // workspace carve-up (256B aligned)
    char* w = (char*)d_ws;
    size_t o = 0;
    auto take = [&](size_t n) { size_t r = o; o += (n + 255) & ~(size_t)255; return r; };
    _Float16* xh   = (_Float16*)(w + take((size_t)NPAD * D * 2));
    float*    h    = (float*)   (w + take((size_t)NPAD * D * 4));
    float*    acc  = (float*)   (w + take((size_t)NPAD * D * 4));
    _Float16* Wp   = (_Float16*)(w + take((size_t)D * D * 2));
    float*    asb  = (float*)   (w + take((size_t)NPAD * 4));
    float*    adb  = (float*)   (w + take((size_t)NPAD * 4));
    unsigned* mu   = (unsigned*)(w + take((size_t)NPAD * 4));
    float*    sb   = (float*)   (w + take((size_t)NPAD * 4));
    float*    ebuf = (float*)   (w + take((size_t)E_TOT * 4));
    float*    eebf = (float*)   (w + take((size_t)E_TOT * 4));
    float*    cnt  = (float*)   (w + take((size_t)NGRAPH * 4));

    const int gElem  = (NPAD * D + 255) / 256;     // 25008
    const int gGemm  = (NPAD + 127) / 128;         // 391
    const int gNodeW = (N_NODES + 7) / 8;          // 6250
    const int gEdge  = (E_TOT + 255) / 256;        // 3321
    const int gEdgeW = (E_TOT + 7) / 8;            // 106250

    k_embed<<<gElem, 256, 0, stream>>>(node_ids, emb, xh);

    for (int l = 0; l < 3; ++l) {
        k_packW<<<(D * D + 255) / 256, 256, 0, stream>>>(W[l], Wp);
        k_init <<<gElem, 256, 0, stream>>>(acc, mu, sb);
        k_gemm <<<gGemm, 256, 0, stream>>>(xh, Wp, h);
        k_att  <<<gNodeW, 256, 0, stream>>>(h, aS[l], aD[l], asb, adb);
        k_emax <<<gEdge, 256, 0, stream>>>(esrc, edst, asb, adb, ebuf, mu);
        k_esum <<<gEdge, 256, 0, stream>>>(esrc, edst, ebuf, mu, eebf, sb);
        k_aggr <<<gEdgeW, 256, 0, stream>>>(esrc, edst, eebf, sb, h, acc);
        k_finish<<<gElem, 256, 0, stream>>>(acc, B[l], xh, l < 2 ? 1 : 0);
    }

    k_pool_zero<<<(NGRAPH * D + 255) / 256, 256, 0, stream>>>(out, cnt);
    k_pool_sum <<<gNodeW, 256, 0, stream>>>(acc, batch, out, cnt);
    k_pool_div <<<(NGRAPH * D + 255) / 256, 256, 0, stream>>>(out, cnt);
}
